// MultiBondFastAttention_83743272337585
// MI455X (gfx1250) — matrix operand order repaired and checked by reading, not compiled, not run on hardware
//
#include <hip/hip_runtime.h>
#include <hip/hip_bf16.h>

#define NN 196608
#define HH 256
#define NHEAD 8
#define DD 32
#define SS 4096
#define NT16 (NN / 16)   // 12288 row strips

typedef __attribute__((ext_vector_type(16))) __bf16 bf16x16;
typedef __attribute__((ext_vector_type(8)))  float  f32x8;

union AFrag { bf16x16 v; __bf16 e[16]; };

// ---------------------------------------------------------------------------
// K0: pack weights into bf16 WMMA B-fragments.
// B-matrix (32x16 bf16) layout used here (contiguous k): lane L holds column n = L%16,
// K = (L/16)*16 + i for element i of the 16-element lane vector.
// Fragment f for (colTile j, ktile kt) stores element (lane, i) at
// P[f*512 + lane*16 + i] with source W[n*H + k] (since y = x @ W^T).
// ---------------------------------------------------------------------------
__global__ __launch_bounds__(256)
void pack_weights(const float* __restrict__ Wq, const float* __restrict__ Wk,
                  const float* __restrict__ Wv, const float* __restrict__ Wo,
                  const float* __restrict__ Wr,
                  __bf16* __restrict__ WqP, __bf16* __restrict__ WkP,
                  __bf16* __restrict__ WvP, __bf16* __restrict__ WoP,
                  __bf16* __restrict__ WrP) {
    int b = blockIdx.x;      // 0..511 big matrices, 512..513 Wr
    int t = threadIdx.x;
    if (b < 512) {
        int mat  = b >> 7;          // 128 fragments per matrix (16 colTiles x 8 ktiles)
        int frag = b & 127;
        int j  = frag >> 3;
        int kt = frag & 7;
        const float* W = (mat == 0) ? Wq : (mat == 1) ? Wk : (mat == 2) ? Wv : Wo;
        __bf16* P      = (mat == 0) ? WqP : (mat == 1) ? WkP : (mat == 2) ? WvP : WoP;
        #pragma unroll
        for (int u = 0; u < 2; ++u) {
            int f = t + 256 * u;
            int lane = f >> 4, i = f & 15;
            int n = j * 16 + (lane & 15);
            int k = kt * 32 + (lane >> 4) * 16 + i;
            P[(size_t)frag * 512 + f] = (__bf16)W[n * HH + k];
        }
    } else {
        int j = b - 512;            // W_r: 2 col tiles (e=0..15, 16..31), K=32
        #pragma unroll
        for (int u = 0; u < 2; ++u) {
            int f = t + 256 * u;
            int lane = f >> 4, i = f & 15;
            int n = j * 16 + (lane & 15);       // e index
            int k = (lane >> 4) * 16 + i;       // d index
            WrP[(size_t)j * 512 + f] = (__bf16)Wr[n * DD + k];
        }
    }
}

// ---------------------------------------------------------------------------
// Helper: build an A fragment (16x32 bf16) from an LDS row-major strip.
// A-matrix layout used here: lanes 0-15 row M=lane, lanes 16-31 row M=lane-16;
// VGPR p -> K = 2p + (lane>=16 ? 16 : 0) + {0,1} (contiguous, matches packed B)
// ---------------------------------------------------------------------------
#define BUILD_AFRAG(dst, ldsrow, colbase, kh)                                  \
    {                                                                          \
        _Pragma("unroll")                                                      \
        for (int p = 0; p < 8; ++p) {                                          \
            int k = p * 2 + (kh) * 16;                                         \
            (dst).e[2 * p]     = (ldsrow)[(colbase) + k];                      \
            (dst).e[2 * p + 1] = (ldsrow)[(colbase) + k + 1];                  \
        }                                                                      \
    }

// ---------------------------------------------------------------------------
// K1: Q/K/V projections. One block = one 16-row strip, 8 waves handle the
// 48 (proj x colTile) tiles. K=256 -> 8 v_wmma_f32_16x16x32_bf16 per tile.
// ---------------------------------------------------------------------------
__global__ __launch_bounds__(256)
void qkv_kernel(const float* __restrict__ msg,
                const __bf16* __restrict__ WqP, const __bf16* __restrict__ WkP,
                const __bf16* __restrict__ WvP,
                float* __restrict__ Q, float* __restrict__ K, float* __restrict__ V) {
    __shared__ __bf16 ldsA[16][HH + 8];
    int t = threadIdx.x;
    size_t rowBase = (size_t)blockIdx.x * 16;

    #pragma unroll
    for (int r = 0; r < 16; ++r)
        ldsA[r][t] = (__bf16)msg[(rowBase + r) * HH + t];
    __syncthreads();

    int wave = t >> 5, lane = t & 31;
    int ln16 = lane & 15, kh = lane >> 4;

    AFrag a[8];                       // all 8 ktiles, reused across 6 tiles/wave
    #pragma unroll
    for (int kt = 0; kt < 8; ++kt)
        BUILD_AFRAG(a[kt], ldsA[ln16], kt * 32, kh);

    for (int tile = wave; tile < 48; tile += 8) {
        int proj = tile / 16;
        int j    = tile % 16;
        const __bf16* P = (proj == 0) ? WqP : (proj == 1) ? WkP : WvP;
        float* O        = (proj == 0) ? Q   : (proj == 1) ? K   : V;
        f32x8 acc = {};
        #pragma unroll
        for (int kt = 0; kt < 8; ++kt) {
            bf16x16 b = *(const bf16x16*)(P + (size_t)(j * 8 + kt) * 512 + (size_t)lane * 16);
            acc = __builtin_amdgcn_wmma_f32_16x16x32_bf16(false, a[kt].v, false, b,
                                                          (short)0, acc, false, false);
        }
        // C/D layout: lane holds column n = lane&15, VGPR r -> row r + 8*(lane>>4)
        #pragma unroll
        for (int r = 0; r < 8; ++r)
            O[(rowBase + r + kh * 8) * HH + j * 16 + ln16] = acc[r];
    }
}

// ---------------------------------------------------------------------------
// K2: per-molecule segment kernel (atomic-free, ids are sorted).
// Block = molecule, wave = head, lane = feature d. Softmax via wave32 shuffles.
// Pass 1: gq = sum_n softmax_d(q*wa)*q   (kept in registers)
// Pass 2: gk = sum_n softmax_d(p*wb)*p,  p = gq*k
// ---------------------------------------------------------------------------
__device__ __forceinline__ float waveMax32(float x) {
    #pragma unroll
    for (int m = 16; m >= 1; m >>= 1) x = fmaxf(x, __shfl_xor(x, m, 32));
    return x;
}
__device__ __forceinline__ float waveSum32(float x) {
    #pragma unroll
    for (int m = 16; m >= 1; m >>= 1) x += __shfl_xor(x, m, 32);
    return x;
}

__global__ __launch_bounds__(256)
void segment_kernel(const float* __restrict__ Q, const float* __restrict__ K,
                    const float* __restrict__ w_alpha, const float* __restrict__ w_beta,
                    const int* __restrict__ seg, float* __restrict__ gk) {
    int s = blockIdx.x;
    int t = threadIdx.x;
    int d = t & 31;

    int lo = 0, hi = NN;                              // lower_bound(s)
    while (lo < hi) { int mid = (lo + hi) >> 1; if (seg[mid] <  s) lo = mid + 1; else hi = mid; }
    int start = lo;
    lo = start; hi = NN;                              // upper_bound(s)
    while (lo < hi) { int mid = (lo + hi) >> 1; if (seg[mid] <= s) lo = mid + 1; else hi = mid; }
    int end = lo;

    const float scale = 0.17677669529663687f;         // D^-0.5
    float wa = w_alpha[d] * scale;
    float wb = w_beta[d]  * scale;

    float gq = 0.f;
    for (int n = start; n < end; ++n) {
        float x  = Q[(size_t)n * HH + t];
        float ar = x * wa;
        float mx = waveMax32(ar);
        float e  = expf(ar - mx);
        float sm = waveSum32(e);
        gq += (e / sm) * x;
    }
    float gkacc = 0.f;
    for (int n = start; n < end; ++n) {
        float p  = gq * K[(size_t)n * HH + t];
        float br = p * wb;
        float mx = waveMax32(br);
        float e  = expf(br - mx);
        float sm = waveSum32(e);
        gkacc += (e / sm) * p;
    }
    gk[(size_t)s * HH + t] = gkacc;
}

// ---------------------------------------------------------------------------
// K3: kvi = gk[seg]*v, per-head W_r GEMM (one WMMA per (head, eTile)),
// +q residual, ReLU, store bf16 X for the output projection.
// ---------------------------------------------------------------------------
__global__ __launch_bounds__(256)
void wr_kernel(const float* __restrict__ V, const float* __restrict__ gk,
               const float* __restrict__ Q, const int* __restrict__ seg,
               const __bf16* __restrict__ WrP, __bf16* __restrict__ X) {
    __shared__ __bf16 ldsA[16][HH + 8];
    __shared__ int segid[16];
    int t = threadIdx.x;
    size_t rowBase = (size_t)blockIdx.x * 16;

    if (t < 16) segid[t] = seg[rowBase + t];
    __syncthreads();
    #pragma unroll
    for (int r = 0; r < 16; ++r) {
        float kvi = gk[(size_t)segid[r] * HH + t] * V[(rowBase + r) * HH + t];
        ldsA[r][t] = (__bf16)kvi;
    }
    __syncthreads();

    int wave = t >> 5, lane = t & 31;
    int ln16 = lane & 15, kh = lane >> 4;

    #pragma unroll
    for (int u = 0; u < 2; ++u) {
        int h = wave;                 // head
        int j = u;                    // e tile (0: e=0..15, 1: e=16..31)
        AFrag a;
        BUILD_AFRAG(a, ldsA[ln16], h * 32, kh);
        bf16x16 b = *(const bf16x16*)(WrP + (size_t)j * 512 + (size_t)lane * 16);
        f32x8 acc = {};
        acc = __builtin_amdgcn_wmma_f32_16x16x32_bf16(false, a.v, false, b,
                                                      (short)0, acc, false, false);
        #pragma unroll
        for (int r = 0; r < 8; ++r) {
            size_t row = rowBase + r + kh * 8;
            int col = h * 32 + j * 16 + ln16;
            float o = acc[r] + Q[row * HH + col];
            X[row * HH + col] = (__bf16)fmaxf(o, 0.f);
        }
    }
}

// ---------------------------------------------------------------------------
// K4: W_o GEMM + bias + LayerNorm, writing output rows 1..N.
// ---------------------------------------------------------------------------
__global__ __launch_bounds__(256)
void out_kernel(const __bf16* __restrict__ X, const __bf16* __restrict__ WoP,
                const float* __restrict__ b_o, const float* __restrict__ gamma,
                const float* __restrict__ beta_ln, float* __restrict__ out) {
    __shared__ __bf16 ldsA[16][HH + 8];
    __shared__ float  ldsO[16][HH];
    __shared__ float  red[2][16][16];
    __shared__ float  mu[16], rstd[16];
    int t = threadIdx.x;
    size_t rowBase = (size_t)blockIdx.x * 16;

    #pragma unroll
    for (int r = 0; r < 16; ++r)
        ldsA[r][t] = X[(rowBase + r) * HH + t];
    __syncthreads();

    int wave = t >> 5, lane = t & 31;
    int ln16 = lane & 15, kh = lane >> 4;

    AFrag a[8];
    #pragma unroll
    for (int kt = 0; kt < 8; ++kt)
        BUILD_AFRAG(a[kt], ldsA[ln16], kt * 32, kh);

    #pragma unroll
    for (int u = 0; u < 2; ++u) {
        int j = wave * 2 + u;
        f32x8 acc = {};
        #pragma unroll
        for (int kt = 0; kt < 8; ++kt) {
            bf16x16 b = *(const bf16x16*)(WoP + (size_t)(j * 8 + kt) * 512 + (size_t)lane * 16);
            acc = __builtin_amdgcn_wmma_f32_16x16x32_bf16(false, a[kt].v, false, b,
                                                          (short)0, acc, false, false);
        }
        int col = j * 16 + ln16;
        float bias = b_o[col];
        #pragma unroll
        for (int r = 0; r < 8; ++r)
            ldsO[r + kh * 8][col] = acc[r] + bias;
    }
    __syncthreads();

    // LayerNorm: 16 threads per row reduce 256 cols
    int row = t >> 4, idx = t & 15;
    float s = 0.f, ss = 0.f;
    #pragma unroll
    for (int c = 0; c < 16; ++c) {
        float v = ldsO[row][idx * 16 + c];
        s += v; ss += v * v;
    }
    red[0][row][idx] = s; red[1][row][idx] = ss;
    __syncthreads();
    if (idx == 0) {
        float S = 0.f, SQ = 0.f;
        #pragma unroll
        for (int c = 0; c < 16; ++c) { S += red[0][row][c]; SQ += red[1][row][c]; }
        float mean = S * (1.0f / HH);
        float var  = SQ * (1.0f / HH) - mean * mean;
        mu[row]   = mean;
        rstd[row] = rsqrtf(var + 1e-5f);
    }
    __syncthreads();

    float g = gamma[t], bl = beta_ln[t];
    #pragma unroll
    for (int r = 0; r < 16; ++r) {
        float v = (ldsO[r][t] - mu[r]) * rstd[r] * g + bl;
        out[(rowBase + r + 1) * HH + t] = v;     // +1: row 0 is the cached zero row
    }
}

__global__ void zero_row_kernel(float* __restrict__ out) {
    out[threadIdx.x] = 0.f;
}

// ---------------------------------------------------------------------------
extern "C" void kernel_launch(void* const* d_in, const int* in_sizes, int n_in,
                              void* d_out, int out_size, void* d_ws, size_t ws_size,
                              hipStream_t stream) {
    const float* msg    = (const float*)d_in[0];
    const float* Wq     = (const float*)d_in[1];
    const float* Wk     = (const float*)d_in[2];
    const float* Wv     = (const float*)d_in[3];
    const float* Wr     = (const float*)d_in[4];
    const float* w_a    = (const float*)d_in[5];
    const float* w_b    = (const float*)d_in[6];
    const float* Wo     = (const float*)d_in[7];
    const float* b_o    = (const float*)d_in[8];
    const float* gamma  = (const float*)d_in[9];
    const float* betaln = (const float*)d_in[10];
    const int*   segid  = (const int*)d_in[11];
    float* out = (float*)d_out;

    // workspace carve-out (256B aligned slots)
    char* ws = (char*)d_ws;
    size_t off = 0;
    auto carve = [&](size_t bytes) -> void* {
        void* p = ws + off;
        off += (bytes + 255) & ~(size_t)255;
        return p;
    };
    float*  Q   = (float*)carve((size_t)NN * HH * sizeof(float));
    float*  K   = (float*)carve((size_t)NN * HH * sizeof(float));
    float*  V   = (float*)carve((size_t)NN * HH * sizeof(float));
    float*  gk  = (float*)carve((size_t)SS * HH * sizeof(float));
    __bf16* X   = (__bf16*)carve((size_t)NN * HH * sizeof(__bf16));
    __bf16* WqP = (__bf16*)carve((size_t)HH * HH * sizeof(__bf16));
    __bf16* WkP = (__bf16*)carve((size_t)HH * HH * sizeof(__bf16));
    __bf16* WvP = (__bf16*)carve((size_t)HH * HH * sizeof(__bf16));
    __bf16* WoP = (__bf16*)carve((size_t)HH * HH * sizeof(__bf16));
    __bf16* WrP = (__bf16*)carve((size_t)DD * DD * sizeof(__bf16));

    pack_weights<<<514, 256, 0, stream>>>(Wq, Wk, Wv, Wo, Wr, WqP, WkP, WvP, WoP, WrP);
    qkv_kernel<<<NT16, 256, 0, stream>>>(msg, WqP, WkP, WvP, Q, K, V);
    segment_kernel<<<SS, 256, 0, stream>>>(Q, K, w_a, w_b, segid, gk);
    wr_kernel<<<NT16, 256, 0, stream>>>(V, gk, Q, segid, WrP, X);
    out_kernel<<<NT16, 256, 0, stream>>>(X, WoP, b_o, gamma, betaln, out);
    zero_row_kernel<<<1, 256, 0, stream>>>(out);
}
